// AttMat_msg_lstm_3925600109085
// MI455X (gfx1250) — matrix operand order repaired and checked by reading, not compiled, not run on hardware
//
#include <hip/hip_runtime.h>
#include <hip/hip_bf16.h>

#define DEV __device__ __forceinline__

typedef __bf16 bf16_t;
typedef __attribute__((ext_vector_type(16))) __bf16 v16bf;
typedef __attribute__((ext_vector_type(8)))  float  v8f;
typedef __attribute__((ext_vector_type(4)))  unsigned int u32x4;
typedef __attribute__((ext_vector_type(8)))  int i32x8;
typedef __attribute__((ext_vector_type(4)))  int i32x4;

// Problem constants
constexpr int Bb = 8, Ss = 32, Nn = 16, Dd = 512, Cc = 6, Pp = 3;
constexpr int NNc = 256;   // N*N flattened (i,w) columns
constexpr int BSc = 256;   // B*S slices

// ---- workspace layout (bytes) ----
constexpr size_t OFF_W1   = 0;                          // 512x512 bf16
constexpr size_t OFF_WMH  = OFF_W1   + 512ull*512*2;    // 512x512 bf16
constexpr size_t OFF_WME  = OFF_WMH  + 512ull*512*2;    // 512x512 bf16
constexpr size_t OFF_WIH  = OFF_WME  + 512ull*512*2;    // 1536x512 bf16
constexpr size_t OFF_WHH  = OFF_WIH  + 1536ull*512*2;   // 1536x512 bf16
constexpr size_t OFF_WLIH = OFF_WHH  + 1536ull*512*2;   // 2048x512 bf16
constexpr size_t OFF_WLHH = OFF_WLIH + 2048ull*512*2;   // 2048x512 bf16
constexpr size_t OFF_X    = OFF_WLHH + 2048ull*512*2;   // 32*128*512 bf16 (LSTM input, [s][row][d])
constexpr size_t OFF_ME   = OFF_X    + 32ull*128*512*2; // 256 * 256*512 bf16 (me^T per slice, [c][d])

// ---- mp_kernel dynamic LDS layout ----
constexpr int SM_E    = 256 * 512 * 2;        // E^T bf16 [c][d]
constexpr int SM_MSUM = 16 * 512 * 4;         // m_sum^T f32 [n][d]
constexpr int SM_H    = 16 * 512 * 2;         // H^T bf16 [n][d]
constexpr int SM_ADJ  = 256 * 4;
constexpr int SM_GATE = 256 * 4;
constexpr int SMEM_MP = SM_E + SM_MSUM + SM_H + SM_ADJ + SM_GATE; // 313,344 B <= 320KB

// ---- lstm_kernel dynamic LDS layout ----
constexpr int LSM_C   = 16 * 512 * 4;   // c state fp32 [n][j]
constexpr int LSM_H   = 16 * 512 * 2;   // h^T bf16 [n][j]
constexpr int LSM_X   = 16 * 512 * 2;   // xt stage bf16 [n][d] (async-copied per step)
constexpr int LSM_LAB = 16 * 8 * 4;
constexpr int SMEM_LSTM = LSM_C + LSM_H + LSM_X + LSM_LAB;  // 66,048 B

DEV float sigmoidf(float x) { return 1.f / (1.f + __expf(-x)); }

union FragBF { v16bf v; u32x4 q[2]; };
union Pack8  { bf16_t h[8]; u32x4 q; };

// A fragment: 16x32 bf16 tile from a row-major (M x K, ld elems/row) source.
// lane<16: row=lane, K 0..15; lane>=16: row=lane-16,
// K 16..31.  => two 16B loads per lane.
DEV v16bf load_a_rowmajor(const bf16_t* __restrict__ base, int ld, int lane) {
  int m = lane & 15, half = lane >> 4;
  const bf16_t* p = base + (size_t)m * ld + half * 16;
  FragBF f;
  f.q[0] = *(const u32x4*)(p);
  f.q[1] = *(const u32x4*)(p + 8);
  return f.v;
}

// B fragment: 32x16 bf16 tile from an N-major (B^T row-major, [n][k]) source.
// lanes0-15 hold K=0..15 of column n=lane, lanes16-31 hold K=16..31. => 32B/lane.
DEV v16bf load_b_nmajor(const bf16_t* __restrict__ base, int ld, int lane) {
  int n = lane & 15, kb = (lane >> 4) * 16;
  const bf16_t* p = base + (size_t)n * ld + kb;
  FragBF f;
  f.q[0] = *(const u32x4*)(p);
  f.q[1] = *(const u32x4*)(p + 8);
  return f.v;
}

// B fragment from an N-major fp32 source, converting to bf16 in registers.
DEV v16bf load_b_nmajor_f32(const float* __restrict__ base, int ld, int lane) {
  int n = lane & 15, kb = (lane >> 4) * 16;
  const float* p = base + (size_t)n * ld + kb;
  v16bf f;
#pragma unroll
  for (int i = 0; i < 16; ++i) f[i] = (bf16_t)p[i];
  return f;
}

DEV v8f wmma_bf16(v16bf a, v16bf b, v8f c) {
  return __builtin_amdgcn_wmma_f32_16x16x32_bf16(false, a, false, b, (short)0, c, false, false);
}

// CDNA5 async global->LDS copy (ASYNCcnt path). 16B per lane per issue.
DEV void async_ld_lds_b128(unsigned lds_byte_addr, const void* gaddr) {
  asm volatile("global_load_async_to_lds_b128 %0, %1, off"
               :: "v"(lds_byte_addr), "v"(gaddr) : "memory");
}
DEV void wait_asynccnt0() { asm volatile("s_wait_asynccnt 0x0" ::: "memory"); }

// ---------------------------------------------------------------------------
__global__ void convert_weights(const float* __restrict__ W1, const float* __restrict__ Wm,
                                const float* __restrict__ Wih, const float* __restrict__ Whh,
                                const float* __restrict__ Wlih, const float* __restrict__ Wlhh,
                                bf16_t* wW1, bf16_t* wWmH, bf16_t* wWmE,
                                bf16_t* wWih, bf16_t* wWhh, bf16_t* wWlih, bf16_t* wWlhh) {
  size_t i0 = (size_t)blockIdx.x * blockDim.x + threadIdx.x;
  size_t st = (size_t)gridDim.x * blockDim.x;
  for (size_t t = i0; t < 512ull * 512; t += st) wW1[t] = (bf16_t)W1[t];
  for (size_t t = i0; t < 512ull * 512; t += st) {
    size_t m = t >> 9, d = t & 511;
    wWmH[t] = (bf16_t)Wm[m * 1024 + d];        // Wm[:, :D]
    wWmE[t] = (bf16_t)Wm[m * 1024 + 512 + d];  // Wm[:, D:]
  }
  for (size_t t = i0; t < 1536ull * 512; t += st) { wWih[t] = (bf16_t)Wih[t]; wWhh[t] = (bf16_t)Whh[t]; }
  for (size_t t = i0; t < 2048ull * 512; t += st) { wWlih[t] = (bf16_t)Wlih[t]; wWlhh[t] = (bf16_t)Wlhh[t]; }
}

// ---------------------------------------------------------------------------
// One workgroup per (b,s) slice.  16 waves (512 threads), wave32.
__global__ void __launch_bounds__(512, 1)
mp_kernel(const float* __restrict__ nodeR, const float* __restrict__ edgeR,
          const int* __restrict__ nodeNum,
          const bf16_t* __restrict__ wW1, const float* __restrict__ b1,
          const float* __restrict__ W2, const float* __restrict__ b2v,
          const bf16_t* __restrict__ wWmH, const bf16_t* __restrict__ wWmE,
          const float* __restrict__ bm,
          const bf16_t* __restrict__ wWih, const bf16_t* __restrict__ wWhh,
          const float* __restrict__ bih, const float* __restrict__ bhh,
          bf16_t* __restrict__ wsMe, bf16_t* __restrict__ wsX,
          float* __restrict__ outAdj) {
  extern __shared__ __align__(16) char smem[];
  bf16_t* sE    = (bf16_t*)(smem);                                   // [256][512]
  float*  sMsum = (float*)(smem + SM_E);                             // [16][512]
  bf16_t* sH    = (bf16_t*)(smem + SM_E + SM_MSUM);                  // [16][512]
  float*  sAdj  = (float*)(smem + SM_E + SM_MSUM + SM_H);            // [256]
  float*  sGate = sAdj + 256;                                        // [256]

  const int bs   = blockIdx.x;        // b*32+s
  const int tid  = threadIdx.x;
  const int lane = tid & 31;
  const int wid  = tid >> 5;          // 0..15
  const int nn   = nodeNum[bs];

  // ---- stage masked edge slice into LDS as bf16 E^T, masked nodes as H^T ----
  const float* eg = edgeR + (size_t)bs * Dd * NNc;
  for (int idx = tid; idx < Dd * NNc; idx += 512) {
    int d = idx >> 8, c = idx & 255;
    int i = c >> 4, w = c & 15;
    bool keep = (i < nn) && (w < nn) && (i != w);
    sE[(size_t)c * Dd + d] = (bf16_t)(keep ? eg[idx] : 0.f);
  }
  const float* ng = nodeR + (size_t)bs * Dd * Nn;
  for (int idx = tid; idx < Dd * Nn; idx += 512) {
    int d = idx >> 4, n = idx & 15;
    sH[(size_t)n * Dd + d] = (bf16_t)((n < nn) ? ng[idx] : 0.f);
  }
  __syncthreads();

  // ---- me = Wm_e @ E0, stored to global ws as me^T [c][d] bf16 ----
  bf16_t* meT = wsMe + (size_t)bs * Dd * NNc;
  for (int job = wid; job < 512; job += 16) {
    int mt = job >> 4, ct = job & 15;
    v8f acc = {};
    for (int k = 0; k < Dd; k += 32) {
      v16bf a = load_a_rowmajor(wWmE + (size_t)(mt * 16) * Dd + k, Dd, lane);
      v16bf b = load_b_nmajor(sE + (size_t)(ct * 16) * Dd + k, Dd, lane);
      acc = wmma_bf16(a, b, acc);
    }
    int n = lane & 15, mrow0 = mt * 16 + (lane >> 4) * 8;
    Pack8 t;
#pragma unroll
    for (int r = 0; r < 8; ++r) t.h[r] = (bf16_t)acc[r];
    *(u32x4*)(meT + (size_t)(ct * 16 + n) * Dd + mrow0) = t.q;
  }
  __threadfence_block();  // make global me visible to the whole workgroup
  __syncthreads();

  // ---- P message-passing rounds ----
  for (int it = 0; it < Pp; ++it) {
    for (int i2 = tid; i2 < 256; i2 += 512) sAdj[i2] = 0.f;
    __syncthreads();

    // adj phase: relu(W1@E + b1) reduced by W2 on the fly
    for (int job = wid; job < 512; job += 16) {
      int ht = job >> 4, ct = job & 15;
      v8f acc = {};
      for (int k = 0; k < Dd; k += 32) {
        __builtin_prefetch(wW1 + (size_t)(ht * 16) * Dd + k + 1024, 0, 1);
        v16bf a = load_a_rowmajor(wW1 + (size_t)(ht * 16) * Dd + k, Dd, lane);
        v16bf b = load_b_nmajor(sE + (size_t)(ct * 16) * Dd + k, Dd, lane);
        acc = wmma_bf16(a, b, acc);
      }
      int n = lane & 15, mo = (lane >> 4) * 8;
      float part = 0.f;
#pragma unroll
      for (int r = 0; r < 8; ++r) {
        int h = ht * 16 + mo + r;
        float v = acc[r] + b1[h];
        v = v > 0.f ? v : 0.f;
        part += v * W2[h];
      }
      atomicAdd(&sAdj[ct * 16 + n], part);  // ds_add_f32
    }
    __syncthreads();

    // gate (and final-iteration sigmoid(adj) output)
    for (int c = tid; c < 256; c += 512) {
      int i = c >> 4, w = c & 15;
      bool em = (i < nn) && (w < nn);
      float adj = em ? (sAdj[c] + b2v[0]) : 0.f;
      sGate[c] = em ? sigmoidf(adj) : 0.f;
      if (it == Pp - 1) outAdj[bs * 256 + c] = sigmoidf(adj);
    }
    __syncthreads();

    // mh + msg phase: each wave owns d-tiles {wid, wid+16}; msg overwrites E in place
    for (int dt = wid; dt < 32; dt += 16) {
      v8f mh = {};
      for (int k = 0; k < Dd; k += 32) {
        v16bf a = load_a_rowmajor(wWmH + (size_t)(dt * 16) * Dd + k, Dd, lane);
        v16bf b = load_b_nmajor(sH + k, Dd, lane);
        mh = wmma_bf16(a, b, mh);
      }
      int w = lane & 15, mo = (lane >> 4) * 8;
      float mhv[8];
#pragma unroll
      for (int r = 0; r < 8; ++r) mhv[r] = mh[r] + bm[dt * 16 + mo + r];

      for (int i = 0; i < 16; ++i) {
        Pack8 me8;
        me8.q = *(const u32x4*)(meT + (size_t)(i * 16 + w) * Dd + dt * 16 + mo);
        float g = sGate[i * 16 + w];
        Pack8 enew;
#pragma unroll
        for (int r = 0; r < 8; ++r) {
          float m = mhv[r] + (float)me8.h[r];
          m = m > 0.f ? m : 0.f;
          m *= g;                           // zero outside emask => E_next = msg
          enew.h[r] = (bf16_t)m;
          // m_sum over w: reduce across the 16 lanes of each half-wave
          float s = m;
          s += __shfl_xor(s, 1);
          s += __shfl_xor(s, 2);
          s += __shfl_xor(s, 4);
          s += __shfl_xor(s, 8);
          if (w == 0) sMsum[(size_t)i * Dd + dt * 16 + mo + r] = s;
        }
        *(u32x4*)(sE + (size_t)(i * 16 + w) * Dd + dt * 16 + mo) = enew.q;
      }
    }
    __syncthreads();

    // GRU phase: 6 WMMA chains per d-tile (ir,iz,in from m_sum; hr,hz,hn from H)
    float hnew[2][8];
    int tc = 0;
    for (int dt = wid; dt < 32; dt += 16, ++tc) {
      v8f air = {}, aiz = {}, ain = {}, ahr = {}, ahz = {}, ahn = {};
      for (int k = 0; k < Dd; k += 32) {
        v16bf bmv = load_b_nmajor_f32(sMsum + k, Dd, lane);
        v16bf bhv = load_b_nmajor(sH + k, Dd, lane);
        v16bf a;
        a = load_a_rowmajor(wWih + (size_t)(dt * 16) * Dd + k, Dd, lane);            air = wmma_bf16(a, bmv, air);
        a = load_a_rowmajor(wWih + (size_t)(512 + dt * 16) * Dd + k, Dd, lane);      aiz = wmma_bf16(a, bmv, aiz);
        a = load_a_rowmajor(wWih + (size_t)(1024 + dt * 16) * Dd + k, Dd, lane);     ain = wmma_bf16(a, bmv, ain);
        a = load_a_rowmajor(wWhh + (size_t)(dt * 16) * Dd + k, Dd, lane);            ahr = wmma_bf16(a, bhv, ahr);
        a = load_a_rowmajor(wWhh + (size_t)(512 + dt * 16) * Dd + k, Dd, lane);      ahz = wmma_bf16(a, bhv, ahz);
        a = load_a_rowmajor(wWhh + (size_t)(1024 + dt * 16) * Dd + k, Dd, lane);     ahn = wmma_bf16(a, bhv, ahn);
      }
      int n = lane & 15, mo = (lane >> 4) * 8;
#pragma unroll
      for (int r = 0; r < 8; ++r) {
        int d = dt * 16 + mo + r;
        float rg = sigmoidf((air[r] + bih[d]) + (ahr[r] + bhh[d]));
        float zg = sigmoidf((aiz[r] + bih[512 + d]) + (ahz[r] + bhh[512 + d]));
        float ngt = tanhf((ain[r] + bih[1024 + d]) + rg * (ahn[r] + bhh[1024 + d]));
        float hold = (float)sH[(size_t)n * Dd + d];
        float hv = (1.f - zg) * ngt + zg * hold;
        hnew[tc][r] = (n < nn) ? hv : 0.f;   // node_resnet masked == 0 off-mask
      }
    }
    __syncthreads();   // everyone done reading old H
    tc = 0;
    for (int dt = wid; dt < 32; dt += 16, ++tc) {
      int n = lane & 15, mo = (lane >> 4) * 8;
      Pack8 hq;
#pragma unroll
      for (int r = 0; r < 8; ++r) hq.h[r] = (bf16_t)hnew[tc][r];
      *(u32x4*)(sH + (size_t)n * Dd + dt * 16 + mo) = hq.q;
    }
    __syncthreads();
  }

  // ---- export node_state (16KB contiguous LDS->global) for LSTM input ----
  int b = bs >> 5, s = bs & 31;
  bf16_t* xrow = wsX + (size_t)(s * 128 + b * 16) * Dd;
#if __has_builtin(__builtin_amdgcn_tensor_store_from_lds) && __has_builtin(__builtin_amdgcn_s_wait_tensorcnt)
  // Tensor Data Mover: 1-D descriptor, data_size=8B, 2048 elements = 16KB.
  if (wid == 0) {
    unsigned long long ga = (unsigned long long)(size_t)xrow;
    unsigned lds = (unsigned)(size_t)sH;   // low 32 bits of flat shared addr = LDS offset
    u32x4 g0;
    g0[0] = 1u;                                              // count=1, is_restore=0
    g0[1] = lds;                                             // lds_addr
    g0[2] = (unsigned)(ga & 0xffffffffu);                    // global_addr[31:0]
    g0[3] = (unsigned)((ga >> 32) & 0x01ffffffu) | (2u << 30); // global_addr[56:32] | type=2
    i32x8 g1;
    g1[0] = (int)(3u << 16);        // workgroup_mask=0, data_size=3 (8B)
    g1[1] = (int)(2048u << 16);     // tensor_dim0 = 2048 (bits 79:48)
    g1[2] = (int)(1u << 16);        // tensor_dim0 hi / tensor_dim1 = 1
    g1[3] = (int)(2048u << 16);     // tile_dim0 = 2048 (bits 127:112)
    g1[4] = 1;                      // tile_dim1 = 1
    g1[5] = 2048;                   // tensor_dim0_stride = 2048
    g1[6] = 0;
    g1[7] = 0;
    i32x4 z4 = {0, 0, 0, 0};
    i32x8 z8 = {0, 0, 0, 0, 0, 0, 0, 0};
    __builtin_amdgcn_tensor_store_from_lds(g0, g1, z4, z4, z8, 0);
    __builtin_amdgcn_s_wait_tensorcnt(0);
  }
#else
  {
    unsigned int* xr = (unsigned int*)xrow;
    const unsigned int* hsrc = (const unsigned int*)sH;
    for (int idx = tid; idx < 16 * Dd / 2; idx += 512) xr[idx] = hsrc[idx];
  }
#endif
}

// ---------------------------------------------------------------------------
// One workgroup per batch b (rows b*16..b*16+15 of the 128-row LSTM state).
__global__ void __launch_bounds__(512, 1)
lstm_kernel(const bf16_t* __restrict__ wsX,
            const bf16_t* __restrict__ wWlih, const bf16_t* __restrict__ wWlhh,
            const float* __restrict__ blih, const float* __restrict__ blhh,
            const float* __restrict__ Wr, const float* __restrict__ br,
            const int* __restrict__ nodeNum, float* __restrict__ outLab) {
  extern __shared__ __align__(16) char smem[];
  float*  sC   = (float*)(smem);                       // [16][512] cell state
  bf16_t* sHb  = (bf16_t*)(smem + LSM_C);              // [16][512] h^T bf16
  bf16_t* sX   = (bf16_t*)(smem + LSM_C + LSM_H);      // [16][512] xt stage
  float*  sLab = (float*)(smem + LSM_C + LSM_H + LSM_X);

  const int b    = blockIdx.x;
  const int tid  = threadIdx.x;
  const int lane = tid & 31;
  const int wid  = tid >> 5;

  for (int i = tid; i < 16 * 512; i += 512) { sC[i] = 0.f; sHb[i] = (bf16_t)0.f; }
  for (int i = tid; i < 16 * 8; i += 512) sLab[i] = 0.f;
  __syncthreads();

  for (int s = 0; s < Ss; ++s) {
    // ---- async-stage xt (16KB) into LDS: global_load_async_to_lds_b128 ----
    const bf16_t* xt = wsX + (size_t)(s * 128 + b * 16) * Dd;
    for (int idx = tid; idx < 16 * 512 / 8; idx += 512) {       // 1024 x 16B
      async_ld_lds_b128((unsigned)(size_t)(sX + idx * 8), (const void*)(xt + idx * 8));
    }
    wait_asynccnt0();
    __syncthreads();

    float hsv[2][8];
    int tc = 0;
    for (int jt = wid; jt < 32; jt += 16, ++tc) {      // hidden 16-tiles
      v8f gacc[4] = {v8f{}, v8f{}, v8f{}, v8f{}};      // i, f, g, o gate tiles
      for (int k = 0; k < Dd; k += 32) {
        v16bf bx = load_b_nmajor(sX + k, Dd, lane);
        v16bf bh = load_b_nmajor(sHb + k, Dd, lane);
#pragma unroll
        for (int q = 0; q < 4; ++q) {
          v16bf a;
          a = load_a_rowmajor(wWlih + (size_t)(q * 512 + jt * 16) * Dd + k, Dd, lane);
          gacc[q] = wmma_bf16(a, bx, gacc[q]);
          a = load_a_rowmajor(wWlhh + (size_t)(q * 512 + jt * 16) * Dd + k, Dd, lane);
          gacc[q] = wmma_bf16(a, bh, gacc[q]);
        }
      }
      int n = lane & 15, mo = (lane >> 4) * 8;
      float lab[Cc];
#pragma unroll
      for (int cidx = 0; cidx < Cc; ++cidx) lab[cidx] = 0.f;
#pragma unroll
      for (int r = 0; r < 8; ++r) {
        int j = jt * 16 + mo + r;
        float gi = gacc[0][r] + blih[j]        + blhh[j];
        float gf = gacc[1][r] + blih[512 + j]  + blhh[512 + j];
        float gg = gacc[2][r] + blih[1024 + j] + blhh[1024 + j];
        float go = gacc[3][r] + blih[1536 + j] + blhh[1536 + j];
        float cv = sigmoidf(gf) * sC[(size_t)n * 512 + j] + sigmoidf(gi) * tanhf(gg);
        sC[(size_t)n * 512 + j] = cv;
        float hv = sigmoidf(go) * tanhf(cv);
        hsv[tc][r] = hv;
#pragma unroll
        for (int cidx = 0; cidx < Cc; ++cidx) lab[cidx] += hv * Wr[cidx * 512 + j];
      }
#pragma unroll
      for (int cidx = 0; cidx < Cc; ++cidx) atomicAdd(&sLab[n * 8 + cidx], lab[cidx]);
    }
    __syncthreads();   // all waves done reading old h / sX
    tc = 0;
    for (int jt = wid; jt < 32; jt += 16, ++tc) {
      int n = lane & 15, mo = (lane >> 4) * 8;
      Pack8 hq;
#pragma unroll
      for (int r = 0; r < 8; ++r) hq.h[r] = (bf16_t)hsv[tc][r];
      *(u32x4*)(sHb + (size_t)n * 512 + jt * 16 + mo) = hq.q;
    }
    __syncthreads();   // sLab fully accumulated
    if (tid < 16 * Cc) {
      int n = tid / Cc, cidx = tid % Cc;
      int nn = nodeNum[b * 32 + s];
      float v = (sLab[n * 8 + cidx] + br[cidx]) * ((n < nn) ? 1.f : 0.f);
      outLab[((size_t)(b * 32 + s) * 16 + n) * Cc + cidx] = v;
      sLab[n * 8 + cidx] = 0.f;
    }
    __syncthreads();
  }
}

// ---------------------------------------------------------------------------
extern "C" void kernel_launch(void* const* d_in, const int* in_sizes, int n_in,
                              void* d_out, int out_size, void* d_ws, size_t ws_size,
                              hipStream_t stream) {
  const float* nodeR = (const float*)d_in[0];
  const float* edgeR = (const float*)d_in[1];
  const int*   nnum  = (const int*)d_in[2];
  const float* W1    = (const float*)d_in[3];
  const float* b1    = (const float*)d_in[4];
  const float* W2    = (const float*)d_in[5];
  const float* b2    = (const float*)d_in[6];
  const float* Wm    = (const float*)d_in[7];
  const float* bm    = (const float*)d_in[8];
  const float* Wih   = (const float*)d_in[9];
  const float* Whh   = (const float*)d_in[10];
  const float* bih   = (const float*)d_in[11];
  const float* bhh   = (const float*)d_in[12];
  const float* Wlih  = (const float*)d_in[13];
  const float* Wlhh  = (const float*)d_in[14];
  const float* blih  = (const float*)d_in[15];
  const float* blhh  = (const float*)d_in[16];
  const float* Wr    = (const float*)d_in[17];
  const float* br    = (const float*)d_in[18];

  char* ws = (char*)d_ws;
  bf16_t* wW1   = (bf16_t*)(ws + OFF_W1);
  bf16_t* wWmH  = (bf16_t*)(ws + OFF_WMH);
  bf16_t* wWmE  = (bf16_t*)(ws + OFF_WME);
  bf16_t* wWih  = (bf16_t*)(ws + OFF_WIH);
  bf16_t* wWhh  = (bf16_t*)(ws + OFF_WHH);
  bf16_t* wWlih = (bf16_t*)(ws + OFF_WLIH);
  bf16_t* wWlhh = (bf16_t*)(ws + OFF_WLHH);
  bf16_t* wsX   = (bf16_t*)(ws + OFF_X);
  bf16_t* wsMe  = (bf16_t*)(ws + OFF_ME);

  float* outAdj = (float*)d_out;                      // (B,S,N,N) = 65536 floats
  float* outLab = (float*)d_out + Bb * Ss * Nn * Nn;  // (B,S,N,C)

  (void)hipFuncSetAttribute(reinterpret_cast<const void*>(mp_kernel),
                            hipFuncAttributeMaxDynamicSharedMemorySize, SMEM_MP);
  (void)hipFuncSetAttribute(reinterpret_cast<const void*>(lstm_kernel),
                            hipFuncAttributeMaxDynamicSharedMemorySize, SMEM_LSTM);

  convert_weights<<<512, 256, 0, stream>>>(W1, Wm, Wih, Whh, Wlih, Wlhh,
                                           wW1, wWmH, wWmE, wWih, wWhh, wWlih, wWlhh);

  mp_kernel<<<BSc, 512, SMEM_MP, stream>>>(nodeR, edgeR, nnum,
                                           wW1, b1, W2, b2, wWmH, wWmE, bm,
                                           wWih, wWhh, bih, bhh,
                                           wsMe, wsX, outAdj);

  lstm_kernel<<<Bb, 512, SMEM_LSTM, stream>>>(wsX, wWlih, wWlhh, blih, blhh,
                                              Wr, br, nnum, outLab);
}
